// TiSASRec_81535659147511
// MI455X (gfx1250) — matrix operand order repaired and checked by reading, not compiled, not run on hardware
//
#include <hip/hip_runtime.h>
#include <hip/hip_bf16.h>

#define B_  16
#define L_  200
#define H_  128
#define NH_ 4
#define NB_ 2
#define HS_ 32
#define LP_ 208          // L padded to 13*16
#define KP_ 224          // attention K dim padded to 7*32 for output GEMM
#define NT_ 257          // time table rows (TSPAN+1)
#define ML_ (B_*L_)      // 3200 rows

typedef __attribute__((ext_vector_type(16))) _Float16 v16h;
typedef __attribute__((ext_vector_type(8)))  _Float16 v8h;
typedef __attribute__((ext_vector_type(8)))  float    v8f;
typedef __attribute__((ext_vector_type(4)))  unsigned int v4u;
typedef __attribute__((ext_vector_type(8)))  int v8i;
typedef __attribute__((ext_vector_type(4)))  int v4i;

#define NEGV (-4294967295.0f)

#if defined(__has_builtin)
#if __has_builtin(__builtin_amdgcn_tensor_load_to_lds) && __has_builtin(__builtin_amdgcn_s_wait_tensorcnt)
#define USE_TDM 1
#endif
#endif

__device__ __forceinline__ v8f wmma32f16(v16h a, v16h b, v8f c) {
  // D = A(16x32 f16) x B(32x16 f16) + C(16x16 f32)
  return __builtin_amdgcn_wmma_f32_16x16x32_f16(false, a, false, b, (short)0, c, false, false);
}

// A-fragment: lane<16 holds A[m][kk+0..15]; lane>=16 holds A[m][kk+16..31].
// Caller passes p = rowbase + kk + (hi?16:0); we read p..p+15.
__device__ __forceinline__ v16h mk_afrag(const _Float16* p) {
  return *(const v16h*)p;
}

__device__ __forceinline__ float wsum(float v) {
#pragma unroll
  for (int o = 16; o > 0; o >>= 1) v += __shfl_xor(v, o, 32);
  return v;
}
__device__ __forceinline__ float wmaxr(float v) {
#pragma unroll
  for (int o = 16; o > 0; o >>= 1) v = fmaxf(v, __shfl_xor(v, o, 32));
  return v;
}

#ifdef USE_TDM
// 2D TDM load of f16 data: `rows` rows of `elems` f16 elements, global row
// stride `stride_elems`, packed contiguously into LDS at byte offset lds_off.
// Descriptor bitfields per CDNA5 ISA section 8.3/8.4 (D# groups 0/1; groups 2/3
// unused for 2D tiles -> zeros). This toolchain's builtin is the 6-arg form:
// (uint32x4 g0, int32x8 g1, int32x4, int32x4, int32x8, i32 cpol).
__device__ __forceinline__ void tdm_load_2d_f16(unsigned lds_off, const void* gaddr,
                                                unsigned elems, unsigned rows,
                                                unsigned stride_elems) {
  unsigned long long ga = (unsigned long long)(size_t)gaddr;
  v4u g0;
  g0[0] = 1u;                                         // count=1 (valid), user mode
  g0[1] = lds_off;                                    // lds_addr (bytes)
  g0[2] = (unsigned)(ga & 0xffffffffu);               // global_addr[31:0]
  g0[3] = (unsigned)((ga >> 32) & 0x01ffffffu) | (2u << 30);  // addr[56:32] | type=2
  v8i g1;
  g1[0] = (int)(1u << 16);                            // wg_mask=0, data_size=1 (2B)
  g1[1] = (int)((elems & 0xffffu) << 16);             // tensor_dim0[15:0]
  g1[2] = (int)(((elems >> 16) & 0xffffu) | ((rows & 0xffffu) << 16));  // dim0 hi | tensor_dim1 lo
  g1[3] = (int)(((rows >> 16) & 0xffffu) | ((elems & 0xffffu) << 16));  // dim1 hi | tile_dim0
  g1[4] = (int)(rows & 0xffffu);                      // tile_dim1 (tile_dim2=0)
  g1[5] = (int)stride_elems;                          // tensor_dim0_stride[31:0]
  g1[6] = 0;
  g1[7] = 0;
  v4i z4 = {0, 0, 0, 0};
  v8i z8 = {0, 0, 0, 0, 0, 0, 0, 0};
  __builtin_amdgcn_tensor_load_to_lds(g0, g1, z4, z4, z8, 0);
}
#endif

// ---------------- elementwise / prep ----------------

__global__ void k_convert(const float* __restrict__ s, _Float16* __restrict__ d, int n) {
  int i = blockIdx.x * blockDim.x + threadIdx.x;
  if (i < n) d[i] = (_Float16)s[i];
}

__global__ void k_embed(const float* __restrict__ emb, const int* __restrict__ seq,
                        float* __restrict__ x, _Float16* __restrict__ xh) {
  int i = blockIdx.x * blockDim.x + threadIdx.x;  // over ML_*H_
  if (i >= ML_ * H_) return;
  int row = i >> 7, col = i & 127;
  int idx = seq[row];
  float v = (idx == 0) ? 0.0f : emb[idx * H_ + col] * 11.313708498984761f;  // sqrt(128)
  x[i] = v; xh[i] = (_Float16)v;
}

// LayerNorm over width 128, one wave per row. Optional second input added first.
__global__ void k_layernorm(const float* __restrict__ a, const float* __restrict__ badd,
                            const float* __restrict__ g, const float* __restrict__ bt,
                            float* __restrict__ of, _Float16* __restrict__ oh) {
  int wid = threadIdx.x >> 5, lane = threadIdx.x & 31;
  int row = blockIdx.x * 8 + wid;
  if (row >= ML_) return;
  const float* pa = a + (size_t)row * H_;
  const float* pb = badd ? badd + (size_t)row * H_ : nullptr;
  float v[4];
  float s = 0.f;
#pragma unroll
  for (int i = 0; i < 4; ++i) {
    int j = lane + 32 * i;
    float t = pa[j];
    if (pb) t += pb[j];
    v[i] = t; s += t;
  }
  float mean = wsum(s) * (1.0f / H_);
  float q = 0.f;
#pragma unroll
  for (int i = 0; i < 4; ++i) { float d = v[i] - mean; q += d * d; }
  float var = wsum(q) * (1.0f / H_);
  float inv = rsqrtf(var + 1e-8f);
#pragma unroll
  for (int i = 0; i < 4; ++i) {
    int j = lane + 32 * i;
    float r = (v[i] - mean) * inv * g[j] + bt[j];
    if (of) of[(size_t)row * H_ + j] = r;
    if (oh) oh[(size_t)row * H_ + j] = (_Float16)r;
  }
}

// ---------------- generic M x 128 x 128 GEMM: C = A @ W^T + bias ----------------
// grid = M/16 blocks of 256 threads; each of 8 waves owns one 16-wide N tile.
__global__ void __launch_bounds__(256)
k_gemm(const _Float16* __restrict__ A, const _Float16* __restrict__ W,
       const float* __restrict__ bias, const float* __restrict__ resid,
       const int* __restrict__ padidx,
       float* __restrict__ Cf, _Float16* __restrict__ Ch, int relu) {
  int m0 = blockIdx.x * 16;
  int wid = threadIdx.x >> 5, lane = threadIdx.x & 31;
  int n0 = wid * 16;
  int mA = lane & 15;
  bool hi = lane >= 16;
  v8f acc = {};
  const _Float16* arow = A + (size_t)(m0 + mA) * H_;
  const _Float16* wrow = W + (size_t)(n0 + mA) * H_;   // B[k][n] = W[n][k]
#pragma unroll
  for (int kk = 0; kk < H_; kk += 32) {
    v16h af = mk_afrag(arow + kk + (hi ? 16 : 0));
    v16h bf = *(const v16h*)(wrow + kk + (hi ? 16 : 0));
    acc = wmma32f16(af, bf, acc);
  }
  int n = n0 + (lane & 15);
  float bv = bias[n];
#pragma unroll
  for (int r = 0; r < 8; ++r) {
    int m = m0 + r + (hi ? 8 : 0);
    float v = acc[r] + bv;
    if (relu) v = fmaxf(v, 0.f);
    if (resid) v += resid[(size_t)m * H_ + n];
    if (padidx && padidx[m] == 0) v = 0.f;
    if (Cf) Cf[(size_t)m * H_ + n] = v;
    if (Ch) Ch[(size_t)m * H_ + n] = (_Float16)v;
  }
}

// ---------------- attention scores: one workgroup per (b,h) ----------------
__global__ void __launch_bounds__(256)
k_scores(const _Float16* __restrict__ Q, const _Float16* __restrict__ K,
         const _Float16* __restrict__ pK, const _Float16* __restrict__ tK,
         const int* __restrict__ tmat, const int* __restrict__ seq,
         float* __restrict__ S) {
  __shared__ __align__(32) _Float16 Qs[LP_][HS_];
  __shared__ __align__(32) _Float16 Ks[LP_][HS_];   // K + pos_K
  __shared__ __align__(32) _Float16 Tk[NT_][HS_];
  int b = blockIdx.x / NH_, h = blockIdx.x % NH_;
  int tid = threadIdx.x;

#ifdef USE_TDM
  // Async tensor DMA: per-head 257x32 slice of tm_K (row stride 128 elems) -> LDS.
  if (tid < 32) {
    tdm_load_2d_f16((unsigned)(size_t)&Tk[0][0], (const void*)(tK + (size_t)h * HS_),
                    HS_, NT_, H_);
  }
#else
  for (int i = tid; i < NT_ * HS_; i += 256) {
    int r = i / HS_, d = i % HS_;
    Tk[r][d] = tK[(size_t)r * H_ + h * HS_ + d];
  }
#endif
  // Overlapped with the DMA: cooperative fill of Q and (K + pos_K).
  for (int i = tid; i < LP_ * HS_; i += 256) {
    int r = i / HS_, d = i % HS_;
    _Float16 qv = (_Float16)0.f, kv = (_Float16)0.f;
    if (r < L_) {
      size_t gi = (size_t)(b * L_ + r) * H_ + h * HS_ + d;
      qv = Q[gi];
      kv = (_Float16)((float)K[gi] + (float)pK[(size_t)r * H_ + h * HS_ + d]);
    }
    Qs[r][d] = qv; Ks[r][d] = kv;
  }
#ifdef USE_TDM
  __builtin_amdgcn_s_wait_tensorcnt(0);
#endif
  __syncthreads();

  int wid = tid >> 5, lane = tid & 31;
  int mA = lane & 15;
  bool hi = lane >= 16;
  const float scale = 0.17677669529663687f;  // 1/sqrt(32)
  float* Sbh = S + (size_t)blockIdx.x * LP_ * LP_;
  for (int t = wid; t < 13 * 13; t += 8) {   // uniform per wave
    int q0 = (t / 13) * 16, n0 = (t % 13) * 16;
    v8f acc = {};
    v16h af = mk_afrag(&Qs[q0 + mA][hi ? 16 : 0]);
    v16h bf = *(const v16h*)(&Ks[n0 + mA][hi ? 16 : 0]);
    acc = wmma32f16(af, bf, acc);
#pragma unroll
    for (int r = 0; r < 8; ++r) {
      int q = q0 + r + (hi ? 8 : 0);
      int k = n0 + (lane & 15);
      float s = NEGV;
      if (q < L_ && k < L_) {
        int tt = tmat[((size_t)b * L_ + q) * L_ + k];
        float dot = 0.f;
#pragma unroll
        for (int d = 0; d < HS_; ++d) dot += (float)Qs[q][d] * (float)Tk[tt][d];
        s = (acc[r] + dot) * scale;
        if (k > q || seq[b * L_ + q] == 0) s = NEGV;  // causal + pad-query mask
      }
      Sbh[q * LP_ + k] = s;
    }
  }
}

// ---------------- softmax: one wave per (bh,q) row; writes zero-padded f16 aw ----------------
__global__ void k_softmax(const float* __restrict__ S, _Float16* __restrict__ AW) {
  int wid = threadIdx.x >> 5, lane = threadIdx.x & 31;
  int row = blockIdx.x * 8 + wid;
  if (row >= B_ * NH_ * LP_) return;
  int q = row % LP_;
  _Float16* ar = AW + (size_t)row * KP_;
  if (q >= L_) {
    for (int k = lane; k < KP_; k += 32) ar[k] = (_Float16)0.f;
    return;
  }
  const float* sr = S + (size_t)row * LP_;
  float mx = -3.4e38f;
  for (int k = lane; k < L_; k += 32) mx = fmaxf(mx, sr[k]);
  mx = wmaxr(mx);
  float ev[7];
  float sum = 0.f;
  for (int i = 0, k = lane; k < L_; ++i, k += 32) { ev[i] = expf(sr[k] - mx); sum += ev[i]; }
  sum = wsum(sum);
  float inv = 1.0f / sum;
  for (int i = 0, k = lane; k < L_; ++i, k += 32) ar[k] = (_Float16)(ev[i] * inv);
  for (int k = L_ + lane; k < KP_; k += 32) ar[k] = (_Float16)0.f;
}

// ---------------- attention output: one workgroup per (b,h) ----------------
__global__ void __launch_bounds__(256)
k_attnout(const _Float16* __restrict__ AW, const _Float16* __restrict__ V,
          const _Float16* __restrict__ pV, const _Float16* __restrict__ tV,
          const int* __restrict__ tmat, float* __restrict__ O) {
  __shared__ __align__(32) _Float16 Vt[HS_][KP_];   // Vt[d][k] = V[k][d] + pV[k][d]
  __shared__ __align__(32) _Float16 Tv[NT_][HS_];
  __shared__ float Oacc[LP_][HS_];
  int b = blockIdx.x / NH_, h = blockIdx.x % NH_;
  int tid = threadIdx.x;

#ifdef USE_TDM
  if (tid < 32) {
    tdm_load_2d_f16((unsigned)(size_t)&Tv[0][0], (const void*)(tV + (size_t)h * HS_),
                    HS_, NT_, H_);
  }
#else
  for (int i = tid; i < NT_ * HS_; i += 256) {
    int r = i / HS_, d = i % HS_;
    Tv[r][d] = tV[(size_t)r * H_ + h * HS_ + d];
  }
#endif
  for (int i = tid; i < HS_ * KP_; i += 256) {
    int d = i / KP_, k = i % KP_;
    _Float16 v = (_Float16)0.f;
    if (k < L_)
      v = (_Float16)((float)V[(size_t)(b * L_ + k) * H_ + h * HS_ + d] +
                     (float)pV[(size_t)k * H_ + h * HS_ + d]);
    Vt[d][k] = v;
  }
#ifdef USE_TDM
  __builtin_amdgcn_s_wait_tensorcnt(0);
#endif
  __syncthreads();

  int wid = tid >> 5, lane = tid & 31;
  int mA = lane & 15;
  bool hi = lane >= 16;
  const _Float16* AWbh = AW + (size_t)blockIdx.x * LP_ * KP_;
  for (int t = wid; t < 13 * 2; t += 8) {    // 13 q-tiles x 2 d-tiles
    int q0 = (t >> 1) * 16, n0 = (t & 1) * 16;
    v8f acc = {};
    const _Float16* arow = AWbh + (size_t)(q0 + mA) * KP_;
#pragma unroll
    for (int kk = 0; kk < KP_; kk += 32) {
      v16h af = mk_afrag(arow + kk + (hi ? 16 : 0));
      v16h bf = *(const v16h*)(&Vt[n0 + (lane & 15)][kk + (hi ? 16 : 0)]);
      acc = wmma32f16(af, bf, acc);
    }
#pragma unroll
    for (int r = 0; r < 8; ++r)
      Oacc[q0 + r + (hi ? 8 : 0)][n0 + (lane & 15)] = acc[r];
  }
  __syncthreads();
  // tV gather term; aw[q][k]==0 for k>q (causal), so loop k<=q.
  for (int q = wid; q < L_; q += 8) {
    int d = lane;
    float o = Oacc[q][d];
    const _Float16* ar = AWbh + (size_t)q * KP_;
    const int* tr = tmat + ((size_t)b * L_ + q) * L_;
    __builtin_prefetch((const void*)ar, 0, 0);
    __builtin_prefetch((const void*)tr, 0, 0);
    for (int k = 0; k <= q; ++k) {
      float a = (float)ar[k];
      o += a * (float)Tv[tr[k]][d];
    }
    O[((size_t)b * L_ + q) * H_ + h * HS_ + d] = o;
  }
}

// ---------------- final logits ----------------
__global__ void k_logits(const float* __restrict__ F, const float* __restrict__ emb,
                         const int* __restrict__ pos, const int* __restrict__ neg,
                         float* __restrict__ out) {
  int wid = threadIdx.x >> 5, lane = threadIdx.x & 31;
  int row = blockIdx.x * 8 + wid;
  if (row >= ML_) return;
  int pi = pos[row], ni = neg[row];
  float sp = 0.f, sn = 0.f;
  for (int j = lane; j < H_; j += 32) {
    float f = F[(size_t)row * H_ + j];
    sp += f * emb[(size_t)pi * H_ + j];
    sn += f * emb[(size_t)ni * H_ + j];
  }
  sp = wsum(sp); sn = wsum(sn);
  if (lane == 0) { out[row] = sp; out[ML_ + row] = sn; }
}

// ---------------- host ----------------
extern "C" void kernel_launch(void* const* d_in, const int* in_sizes, int n_in,
                              void* d_out, int out_size, void* d_ws, size_t ws_size,
                              hipStream_t stream) {
  (void)in_sizes; (void)n_in; (void)out_size; (void)ws_size;
  const int*   log_seqs = (const int*)d_in[0];
  const int*   tmat     = (const int*)d_in[1];
  const int*   pos_seqs = (const int*)d_in[2];
  const int*   neg_seqs = (const int*)d_in[3];
  const float* item_emb = (const float*)d_in[5];
  const float* pos_K = (const float*)d_in[6];
  const float* pos_V = (const float*)d_in[7];
  const float* tm_K  = (const float*)d_in[8];
  const float* tm_V  = (const float*)d_in[9];
  const float* Wq = (const float*)d_in[10]; const float* bq = (const float*)d_in[11];
  const float* Wk = (const float*)d_in[12]; const float* bk = (const float*)d_in[13];
  const float* Wv = (const float*)d_in[14]; const float* bv = (const float*)d_in[15];
  const float* ln1_g = (const float*)d_in[16]; const float* ln1_b = (const float*)d_in[17];
  const float* ln2_g = (const float*)d_in[18]; const float* ln2_b = (const float*)d_in[19];
  const float* W1 = (const float*)d_in[20]; const float* b1 = (const float*)d_in[21];
  const float* W2 = (const float*)d_in[22]; const float* b2 = (const float*)d_in[23];
  const float* lnf_g = (const float*)d_in[24]; const float* lnf_b = (const float*)d_in[25];
  float* out = (float*)d_out;

  char* base = (char*)d_ws; size_t off = 0;
  auto alloc = [&](size_t bytes) -> void* {
    void* p = base + off; off = (off + bytes + 255) & ~(size_t)255; return p;
  };

  float* x      = (float*)alloc((size_t)ML_ * H_ * 4);
  float* qin    = (float*)alloc((size_t)ML_ * H_ * 4);
  float* attn   = (float*)alloc((size_t)ML_ * H_ * 4);
  float* scores = (float*)alloc((size_t)B_ * NH_ * LP_ * LP_ * 4);
  _Float16* xh   = (_Float16*)alloc((size_t)ML_ * H_ * 2);
  _Float16* qinh = (_Float16*)alloc((size_t)ML_ * H_ * 2);
  _Float16* qh   = (_Float16*)alloc((size_t)ML_ * H_ * 2);
  _Float16* kh   = (_Float16*)alloc((size_t)ML_ * H_ * 2);
  _Float16* vh   = (_Float16*)alloc((size_t)ML_ * H_ * 2);
  _Float16* hm   = (_Float16*)alloc((size_t)ML_ * H_ * 2);
  _Float16* aw   = (_Float16*)alloc((size_t)B_ * NH_ * LP_ * KP_ * 2);
  _Float16* wqh = (_Float16*)alloc((size_t)NB_ * H_ * H_ * 2);
  _Float16* wkh = (_Float16*)alloc((size_t)NB_ * H_ * H_ * 2);
  _Float16* wvh = (_Float16*)alloc((size_t)NB_ * H_ * H_ * 2);
  _Float16* w1h = (_Float16*)alloc((size_t)NB_ * H_ * H_ * 2);
  _Float16* w2h = (_Float16*)alloc((size_t)NB_ * H_ * H_ * 2);
  _Float16* pkh = (_Float16*)alloc((size_t)L_ * H_ * 2);
  _Float16* pvh = (_Float16*)alloc((size_t)L_ * H_ * 2);
  _Float16* tkh = (_Float16*)alloc((size_t)NT_ * H_ * 2);
  _Float16* tvh = (_Float16*)alloc((size_t)NT_ * H_ * 2);

  auto conv = [&](const float* s, _Float16* d, int n) {
    k_convert<<<(n + 255) / 256, 256, 0, stream>>>(s, d, n);
  };
  conv(Wq, wqh, NB_ * H_ * H_); conv(Wk, wkh, NB_ * H_ * H_); conv(Wv, wvh, NB_ * H_ * H_);
  conv(W1, w1h, NB_ * H_ * H_); conv(W2, w2h, NB_ * H_ * H_);
  conv(pos_K, pkh, L_ * H_); conv(pos_V, pvh, L_ * H_);
  conv(tm_K, tkh, NT_ * H_); conv(tm_V, tvh, NT_ * H_);

  k_embed<<<(ML_ * H_ + 255) / 256, 256, 0, stream>>>(item_emb, log_seqs, x, xh);

  for (int i = 0; i < NB_; ++i) {
    const int WO = i * H_ * H_;
    k_layernorm<<<400, 256, 0, stream>>>(x, nullptr, ln1_g + i * H_, ln1_b + i * H_, qin, qinh);
    k_gemm<<<200, 256, 0, stream>>>(qinh, wqh + WO, bq + i * H_, nullptr, nullptr, nullptr, qh, 0);
    k_gemm<<<200, 256, 0, stream>>>(xh,   wkh + WO, bk + i * H_, nullptr, nullptr, nullptr, kh, 0);
    k_gemm<<<200, 256, 0, stream>>>(xh,   wvh + WO, bv + i * H_, nullptr, nullptr, nullptr, vh, 0);
    k_scores<<<B_ * NH_, 256, 0, stream>>>(qh, kh, pkh, tkh, tmat, log_seqs, scores);
    k_softmax<<<(B_ * NH_ * LP_) / 8, 256, 0, stream>>>(scores, aw);
    k_attnout<<<B_ * NH_, 256, 0, stream>>>(aw, vh, pvh, tvh, tmat, attn);
    k_layernorm<<<400, 256, 0, stream>>>(qin, attn, ln2_g + i * H_, ln2_b + i * H_, x, xh);
    k_gemm<<<200, 256, 0, stream>>>(xh, w1h + WO, b1 + i * H_, nullptr, nullptr, nullptr, hm, 1);
    k_gemm<<<200, 256, 0, stream>>>(hm, w2h + WO, b2 + i * H_, x, log_seqs, x, xh, 0);
  }
  k_layernorm<<<400, 256, 0, stream>>>(x, nullptr, lnf_g, lnf_b, qin, nullptr);
  k_logits<<<400, 256, 0, stream>>>(qin, item_emb, pos_seqs, neg_seqs, out);
}
